// HybridQuantLinear_22110491639903
// MI455X (gfx1250) — 8-bit matrix operand order repaired and checked by reading, compiled, not run on hardware
//
#include <hip/hip_runtime.h>
#include <stdint.h>

typedef __attribute__((ext_vector_type(8))) int v8i;
typedef __attribute__((ext_vector_type(4))) int v4i;
typedef __attribute__((ext_vector_type(2))) int v2i;

#define TOKENS 512
#define IN_F   8192
#define OUT_F  8192
#define RANK   64
#define NNZ    8192

#define BM 128
#define BN 128
#define BK 64
#define LDS_STRIDE 80
#define LR_TT 16

__global__ void quant_act_kernel(const float* __restrict__ x,
                                 const float* __restrict__ smooth,
                                 const float* __restrict__ act_scale,
                                 int* __restrict__ xq_packed)
{
    const int idx = blockIdx.x * blockDim.x + threadIdx.x;
    const float inv_a = 1.0f / act_scale[0];
    const size_t base = (size_t)idx * 4;
    const int k = (int)(base & (IN_F - 1));
    int packed = 0;
#pragma unroll
    for (int i = 0; i < 4; ++i) {
        float v = x[base + i] / smooth[k + i];
        float q = rintf(v * inv_a);
        q = fminf(fmaxf(q, -128.0f), 127.0f);
        packed |= ((int)q & 0xFF) << (8 * i);
    }
    xq_packed[idx] = packed;
}

__global__ void pack_w_kernel(const int* __restrict__ qw, int* __restrict__ w8)
{
    const int idx = blockIdx.x * blockDim.x + threadIdx.x;
    const v4i w = ((const v4i*)qw)[idx];
    w8[idx] = (w[0] & 0xFF) | ((w[1] & 0xFF) << 8) |
              ((w[2] & 0xFF) << 16) | ((w[3] & 0xFF) << 24);
}

__launch_bounds__(256)
__global__ void gemm_iu8_kernel(const int8_t* __restrict__ xq,
                                const int8_t* __restrict__ w8,
                                const float* __restrict__ wscale,
                                const float* __restrict__ act_scale,
                                const float* __restrict__ bias,
                                float* __restrict__ y)
{
    __shared__ __align__(16) unsigned char ldsA[2][BM * LDS_STRIDE];
    __shared__ __align__(16) unsigned char ldsB[2][BN * LDS_STRIDE];

    const int tid  = threadIdx.x;
    const int lane = tid & 31;
    const int wave = tid >> 5;
    const int wm   = wave >> 2;
    const int wn   = wave & 3;
    const int l15  = lane & 15;
    const int lh   = lane >> 4;

    const int m0 = blockIdx.y * BM;
    const int n0 = blockIdx.x * BN;

    v8i acc[4][2];
#pragma unroll
    for (int mi = 0; mi < 4; ++mi)
#pragma unroll
        for (int ni = 0; ni < 2; ++ni)
#pragma unroll
            for (int e = 0; e < 8; ++e) acc[mi][ni][e] = 0;

    const int NKB = IN_F / BK;

    const int rowc = tid >> 2;
    const int colc = (tid & 3) * 16;
    unsigned long long gA0 = (unsigned long long)(uintptr_t)(xq + (size_t)(m0 + rowc)      * IN_F + colc);
    unsigned long long gA1 = (unsigned long long)(uintptr_t)(xq + (size_t)(m0 + rowc + 64) * IN_F + colc);
    unsigned long long gB0 = (unsigned long long)(uintptr_t)(w8 + (size_t)(n0 + rowc)      * IN_F + colc);
    unsigned long long gB1 = (unsigned long long)(uintptr_t)(w8 + (size_t)(n0 + rowc + 64) * IN_F + colc);

    auto issue_tile = [&](int buf) {
        unsigned lA0 = (unsigned)(uintptr_t)&ldsA[buf][rowc * LDS_STRIDE + colc];
        unsigned lA1 = (unsigned)(uintptr_t)&ldsA[buf][(rowc + 64) * LDS_STRIDE + colc];
        unsigned lB0 = (unsigned)(uintptr_t)&ldsB[buf][rowc * LDS_STRIDE + colc];
        unsigned lB1 = (unsigned)(uintptr_t)&ldsB[buf][(rowc + 64) * LDS_STRIDE + colc];
        asm volatile("global_load_async_to_lds_b128 %0, %1, off" :: "v"(lA0), "v"(gA0) : "memory");
        asm volatile("global_load_async_to_lds_b128 %0, %1, off" :: "v"(lA1), "v"(gA1) : "memory");
        asm volatile("global_load_async_to_lds_b128 %0, %1, off" :: "v"(lB0), "v"(gB0) : "memory");
        asm volatile("global_load_async_to_lds_b128 %0, %1, off" :: "v"(lB1), "v"(gB1) : "memory");
        gA0 += BK; gA1 += BK; gB0 += BK; gB1 += BK;
    };

    auto compute_tile = [&](int buf) {
        v8i bfrag[2];
#pragma unroll
        for (int ni = 0; ni < 2; ++ni) {
            const unsigned char* p =
                &ldsB[buf][(wn * 32 + ni * 16 + l15) * LDS_STRIDE + lh * 16];
            v4i b0 = *(const v4i*)(p +  0);
            v4i b1 = *(const v4i*)(p + 32);
            bfrag[ni][0] = b0[0]; bfrag[ni][1] = b0[1];
            bfrag[ni][2] = b0[2]; bfrag[ni][3] = b0[3];
            bfrag[ni][4] = b1[0]; bfrag[ni][5] = b1[1];
            bfrag[ni][6] = b1[2]; bfrag[ni][7] = b1[3];
        }
#pragma unroll
        for (int mi = 0; mi < 4; ++mi) {
            const unsigned char* p =
                &ldsA[buf][(wm * 64 + mi * 16 + l15) * LDS_STRIDE + lh * 16];
            v2i q0 = *(const v2i*)(p +  0);
            v2i q1 = *(const v2i*)(p +  8);
            v2i q2 = *(const v2i*)(p + 32);
            v2i q3 = *(const v2i*)(p + 40);
            v8i af;
            af[0] = q0[0]; af[1] = q0[1];
            af[2] = q1[0]; af[3] = q1[1];
            af[4] = q2[0]; af[5] = q2[1];
            af[6] = q3[0]; af[7] = q3[1];
#pragma unroll
            for (int ni = 0; ni < 2; ++ni)
                acc[mi][ni] = __builtin_amdgcn_wmma_i32_16x16x64_iu8(
                    true, af, true, bfrag[ni], acc[mi][ni], false, false);
        }
    };

    issue_tile(0);
    for (int kb = 0; kb < NKB; kb += 2) {
        issue_tile(1);
        asm volatile("s_wait_asynccnt 0x4" ::: "memory");
        __syncthreads();
        compute_tile(0);
        __syncthreads();

        if (kb + 2 < NKB) {
            issue_tile(0);
            asm volatile("s_wait_asynccnt 0x4" ::: "memory");
        } else {
            asm volatile("s_wait_asynccnt 0x0" ::: "memory");
        }
        __syncthreads();
        compute_tile(1);
        __syncthreads();
    }

    const float a_s = act_scale[0];
#pragma unroll
    for (int ni = 0; ni < 2; ++ni) {
        const int oc   = n0 + wn * 32 + ni * 16 + l15;
        const float sc = a_s * wscale[oc];
        const float bz = bias[oc];
#pragma unroll
        for (int mi = 0; mi < 4; ++mi) {
            const int tbase = m0 + wm * 64 + mi * 16 + lh * 8;
#pragma unroll
            for (int v = 0; v < 8; ++v)
                y[(size_t)(tbase + v) * OUT_F + oc] = (float)acc[mi][ni][v] * sc + bz;
        }
    }
}

__global__ void outlier_kernel(const int8_t* __restrict__ xq,
                               const int* __restrict__ rows,
                               const int* __restrict__ cols,
                               const float* __restrict__ vals,
                               const float* __restrict__ act_scale,
                               float* __restrict__ y)
{
    const int n = blockIdx.x;
    const int r = rows[n];
    const int c = cols[n];
    const float v = vals[n] * act_scale[0];
    for (int t = threadIdx.x; t < TOKENS; t += blockDim.x) {
        const float xd = (float)xq[(size_t)t * IN_F + c];
        atomicAdd(&y[(size_t)t * OUT_F + r], xd * v);
    }
}

__global__ void xv_kernel(const int8_t* __restrict__ xq,
                          const float* __restrict__ V,
                          const float* __restrict__ S,
                          const float* __restrict__ act_scale,
                          float* __restrict__ xv)
{
    const int t = blockIdx.x * 4 + (threadIdx.x >> 6);
    const int r = threadIdx.x & 63;
    const int8_t* xr = xq + (size_t)t * IN_F;
    float acc = 0.f;
#pragma unroll 8
    for (int k = 0; k < IN_F; ++k)
        acc += (float)xr[k] * V[(size_t)k * RANK + r];
    xv[t * RANK + r] = acc * act_scale[0] * S[r];
}

__global__ void lowrank_add_kernel(const float* __restrict__ xv,
                                   const float* __restrict__ U,
                                   float* __restrict__ y)
{
    __shared__ float s[LR_TT][RANK];
    const int t0 = blockIdx.y * LR_TT;
    const int o  = blockIdx.x * blockDim.x + threadIdx.x;

    for (int i = threadIdx.x; i < LR_TT * RANK; i += blockDim.x)
        ((float*)s)[i] = xv[t0 * RANK + i];
    __syncthreads();

    const float* u = U + (size_t)o * RANK;
    float acc[LR_TT];
#pragma unroll
    for (int tt = 0; tt < LR_TT; ++tt) acc[tt] = 0.f;

#pragma unroll 8
    for (int r = 0; r < RANK; ++r) {
        const float uv = u[r];
#pragma unroll
        for (int tt = 0; tt < LR_TT; ++tt) acc[tt] += s[tt][r] * uv;
    }
#pragma unroll
    for (int tt = 0; tt < LR_TT; ++tt)
        y[(size_t)(t0 + tt) * OUT_F + o] += acc[tt];
}

extern "C" void kernel_launch(void* const* d_in, const int* in_sizes, int n_in,
                              void* d_out, int out_size, void* d_ws, size_t ws_size,
                              hipStream_t stream)
{
    const float* x       = (const float*)d_in[0];
    const int*   qw      = (const int*)  d_in[1];
    const float* wscale  = (const float*)d_in[2];
    const float* smooth  = (const float*)d_in[3];
    const float* a_scale = (const float*)d_in[4];
    const int*   orow    = (const int*)  d_in[5];
    const int*   ocol    = (const int*)  d_in[6];
    const float* ovals   = (const float*)d_in[7];
    const float* svdU    = (const float*)d_in[8];
    const float* svdS    = (const float*)d_in[9];
    const float* svdV    = (const float*)d_in[10];
    const float* bias    = (const float*)d_in[11];
    float* y = (float*)d_out;

    char* ws = (char*)d_ws;
    int8_t* xq = (int8_t*)ws;
    int8_t* w8 = (int8_t*)(ws + (size_t)TOKENS * IN_F);
    float*  xv = (float*)(ws + (size_t)TOKENS * IN_F + (size_t)OUT_F * IN_F);

    quant_act_kernel<<<TOKENS * IN_F / 4 / 256, 256, 0, stream>>>(x, smooth, a_scale, (int*)xq);
    pack_w_kernel<<<(OUT_F / 256) * (IN_F / 4), 256, 0, stream>>>(qw, (int*)w8);

    dim3 grid(OUT_F / BN, TOKENS / BM);
    gemm_iu8_kernel<<<grid, 256, 0, stream>>>(xq, w8, wscale, a_scale, bias, y);

    outlier_kernel<<<NNZ, 256, 0, stream>>>(xq, orow, ocol, ovals, a_scale, y);
    xv_kernel<<<TOKENS / 4, 256, 0, stream>>>(xq, svdV, svdS, a_scale, xv);
    lowrank_add_kernel<<<dim3(OUT_F / 256, TOKENS / LR_TT), 256, 0, stream>>>(xv, svdU, y);
}
